// Eagle3Attention_71098888618552
// MI455X (gfx1250) — compile-verified, matrix operand order checked by reading, not run on hardware
//
#include <hip/hip_runtime.h>
#include <hip/hip_bf16.h>

// ---------------------------------------------------------------------------
// Eagle3 attention prefill for MI455X (gfx1250, wave32, WMMA).
// All heavy math runs through v_wmma_f32_16x16x32_bf16 (f32->bf16 once,
// f32 accumulate). Compute-bound (~1000 FLOP/byte). GEMM tile staging uses
// CDNA5 async global->LDS DMA (GLOBAL_LOAD_ASYNC_TO_LDS_B128, ASYNCcnt),
// freeing VGPRs and the vector pipe for the WMMA stream.
// ---------------------------------------------------------------------------

typedef __bf16 bf16_t;
typedef __attribute__((ext_vector_type(16))) __bf16 v16bf;
typedef __attribute__((ext_vector_type(8)))  __bf16 v8bf;
typedef __attribute__((ext_vector_type(8)))  float  v8f;

#define SQ      2048
#define SK      4096
#define PAST    2048
#define NH      32
#define NKV     8
#define HD      128
#define DIN     8192
#define QDIM    (NH * HD)    // 4096
#define KVDIM   (NKV * HD)   // 1024

__device__ __forceinline__ v8f wmma_bf16(v16bf a, v16bf b, v8f c) {
  // D = A(16x32 bf16) x B(32x16 bf16) + C(16x16 f32)
  return __builtin_amdgcn_wmma_f32_16x16x32_bf16(
      /*neg_a=*/false, a, /*neg_b=*/false, b,
      /*c_mod=*/(short)0, c, /*reuse_a=*/false, /*reuse_b=*/false);
}

// Load a 16-bit A/B fragment. Both the A matrix and (by symmetry) the B
// matrix use: lane l -> row (row0 + (l&15)); lanes 0-15 hold K {0..7,16..23},
// lanes 16-31 hold K {8..15,24..31} relative to k0.  Two 16-byte loads.
__device__ __forceinline__ v16bf frag16(const bf16_t* __restrict__ base,
                                        int ld, int row0, int k0, int lane) {
  int r  = row0 + (lane & 15);
  int kb = k0 + ((lane & 16) ? 8 : 0);
  const bf16_t* p = base + (size_t)r * ld + kb;
  v8bf lo = *reinterpret_cast<const v8bf*>(p);
  v8bf hi = *reinterpret_cast<const v8bf*>(p + 16);
  return __builtin_shufflevector(lo, hi, 0,1,2,3,4,5,6,7,8,9,10,11,12,13,14,15);
}

// Async 16B/lane global -> LDS DMA (CDNA5 TDM-lite path, ASYNCcnt-tracked).
// VDST = per-lane LDS byte offset (low 32 bits of the generic pointer),
// VADDR = per-lane 64-bit global address.
__device__ __forceinline__ void async_g2l_b128(void* lds_ptr, const void* gptr) {
  unsigned lds = (unsigned)(size_t)lds_ptr;
  unsigned long long g = (unsigned long long)gptr;
  asm volatile("global_load_async_to_lds_b128 %0, %1, off"
               :: "v"(lds), "v"(g) : "memory");
}
__device__ __forceinline__ void wait_async0() {
  asm volatile("s_wait_asynccnt 0x0" ::: "memory");
}

// ---------------------------------------------------------------------------
// f32 -> bf16 bulk convert (n multiple of 8)
// ---------------------------------------------------------------------------
__global__ void f32_to_bf16(const float* __restrict__ in,
                            bf16_t* __restrict__ out, size_t n) {
  size_t i = ((size_t)blockIdx.x * blockDim.x + threadIdx.x) * 8;
  if (i + 8 <= n) {
    float4 a = *reinterpret_cast<const float4*>(in + i);
    float4 b = *reinterpret_cast<const float4*>(in + i + 4);
    v8bf o;
    o[0] = (bf16_t)a.x; o[1] = (bf16_t)a.y; o[2] = (bf16_t)a.z; o[3] = (bf16_t)a.w;
    o[4] = (bf16_t)b.x; o[5] = (bf16_t)b.y; o[6] = (bf16_t)b.z; o[7] = (bf16_t)b.w;
    *reinterpret_cast<v8bf*>(out + i) = o;
  }
}

// ---------------------------------------------------------------------------
// Tiled WMMA GEMM: C[M,N] = A(MxK) * B(NxK)^T + bias.  A,B bf16 row-major
// (K contiguous for both -> identical fragment pattern), C f32.
// Block 256 thr = 8 waves, tile 128Mx128N, BK=64 staged in LDS via async DMA.
// Wave (w&3, w>>2) computes a 32x64 sub-tile = 2x4 WMMA tiles.
// Requires M%128==0, N%128==0, K%64==0.
// ---------------------------------------------------------------------------
#define GBK 64
#define GLD (GBK + 8)   // 72 elem = 144B row stride, keeps 16B alignment

__global__ __launch_bounds__(256) void gemm_bf16(
    const bf16_t* __restrict__ A, const bf16_t* __restrict__ B,
    const float* __restrict__ bias, float* __restrict__ C,
    int M, int N, int K) {
  __shared__ __align__(16) bf16_t As[128][GLD];
  __shared__ __align__(16) bf16_t Bs[128][GLD];

  int tid  = threadIdx.x;
  int lane = tid & 31;
  int w    = tid >> 5;
  int bm   = blockIdx.y * 128;
  int bn   = blockIdx.x * 128;
  int wm   = (w & 3) * 32;
  int wn   = (w >> 2) * 64;

  // This thread's 4 staging chunks per matrix: chunk c -> row c>>3, col (c&7)*8
  int srow[4], scol[4];
#pragma unroll
  for (int i = 0; i < 4; ++i) {
    int c = tid + i * 256;
    srow[i] = c >> 3;
    scol[i] = (c & 7) << 3;
  }

  v8f acc[2][4];
#pragma unroll
  for (int mi = 0; mi < 2; ++mi)
#pragma unroll
    for (int ni = 0; ni < 4; ++ni)
      acc[mi][ni] = (v8f){0.f,0.f,0.f,0.f,0.f,0.f,0.f,0.f};

  for (int k0 = 0; k0 < K; k0 += GBK) {
    // Async-DMA the 128x64 bf16 tiles of A and B straight into LDS
    // (no VGPR staging): 8 x GLOBAL_LOAD_ASYNC_TO_LDS_B128 per thread.
#pragma unroll
    for (int i = 0; i < 4; ++i) {
      async_g2l_b128(&As[srow[i]][scol[i]],
                     A + (size_t)(bm + srow[i]) * K + k0 + scol[i]);
      async_g2l_b128(&Bs[srow[i]][scol[i]],
                     B + (size_t)(bn + srow[i]) * K + k0 + scol[i]);
    }
    wait_async0();       // my lanes' DMA landed in LDS
    __syncthreads();     // everyone's did

    // Warm the cache for the next K tile while the WMMAs run.
    if (k0 + GBK < K) {
#pragma unroll
      for (int i = 0; i < 4; ++i) {
        __builtin_prefetch(A + (size_t)(bm + srow[i]) * K + k0 + GBK + scol[i], 0, 1);
        __builtin_prefetch(B + (size_t)(bn + srow[i]) * K + k0 + GBK + scol[i], 0, 1);
      }
    }

#pragma unroll
    for (int kk = 0; kk < GBK; kk += 32) {
      v16bf af[2], bfm[4];
#pragma unroll
      for (int mi = 0; mi < 2; ++mi)
        af[mi] = frag16(&As[0][0], GLD, wm + mi * 16, kk, lane);
#pragma unroll
      for (int ni = 0; ni < 4; ++ni)
        bfm[ni] = frag16(&Bs[0][0], GLD, wn + ni * 16, kk, lane);
#pragma unroll
      for (int mi = 0; mi < 2; ++mi)
#pragma unroll
        for (int ni = 0; ni < 4; ++ni)
          acc[mi][ni] = wmma_bf16(af[mi], bfm[ni], acc[mi][ni]);
    }
    __syncthreads();
  }

  // D layout: lane&15 = column n; VGPR r = row r (+8 for lanes 16-31).
  int ncol = lane & 15;
  int ra   = (lane & 16) ? 8 : 0;
#pragma unroll
  for (int mi = 0; mi < 2; ++mi)
#pragma unroll
    for (int ni = 0; ni < 4; ++ni) {
      int gn = bn + wn + ni * 16 + ncol;
      float bv = bias ? bias[gn] : 0.f;
#pragma unroll
      for (int r = 0; r < 8; ++r) {
        int gm = bm + wm + mi * 16 + r + ra;
        C[(size_t)gm * N + gn] = acc[mi][ni][r] + bv;
      }
    }
}

// ---------------------------------------------------------------------------
// RoPE on Q and new-K.  hh < 32 -> Q head, hh >= 32 -> KV head (hh-32).
// Writes: Qb bf16 [NH][SQ][HD];  K -> f32 key tuple output at l=PAST+s and
// bf16 Kb [NKV][SK][HD].
// ---------------------------------------------------------------------------
__global__ void rope_qk(const float* __restrict__ Qf,   // [SQ][QDIM]
                        const float* __restrict__ Kf,   // [SQ][KVDIM]
                        const int*   __restrict__ pos,  // [SQ]
                        bf16_t* __restrict__ Qb,
                        float*  __restrict__ keyOut,    // [NKV][SK][HD]
                        bf16_t* __restrict__ Kb) {
  int idx = blockIdx.x * blockDim.x + threadIdx.x;
  int d   = idx & 63;
  int hh  = (idx >> 6) % (NH + NKV);
  int s   = idx / (64 * (NH + NKV));
  if (s >= SQ) return;

  float p   = (float)pos[s];
  // inv_freq = theta^(-2d/HD)
  float inv = __expf((-2.f * (float)d / (float)HD) * 9.210340371976184f);
  float fr  = p * inv;
  float sn, cs;
  __sincosf(fr, &sn, &cs);

  if (hh < NH) {
    const float* q = Qf + (size_t)s * QDIM + hh * HD;
    float x1 = q[d], x2 = q[d + 64];
    bf16_t* o = Qb + ((size_t)hh * SQ + s) * HD;
    o[d]      = (bf16_t)(x1 * cs - x2 * sn);
    o[d + 64] = (bf16_t)(x2 * cs + x1 * sn);
  } else {
    int h = hh - NH;
    const float* k = Kf + (size_t)s * KVDIM + h * HD;
    float x1 = k[d], x2 = k[d + 64];
    float o1 = x1 * cs - x2 * sn;
    float o2 = x2 * cs + x1 * sn;
    size_t l = (size_t)PAST + s;
    size_t base = ((size_t)h * SK + l) * HD;
    keyOut[base + d]      = o1;
    keyOut[base + d + 64] = o2;
    Kb[base + d]      = (bf16_t)o1;
    Kb[base + d + 64] = (bf16_t)o2;
  }
}

// ---------------------------------------------------------------------------
// Cache ingest: copy caches into tuple outputs + bf16 K and transposed bf16 V.
// ---------------------------------------------------------------------------
__global__ void ingest_cache(const float* __restrict__ ck,   // [NKV][PAST][HD]
                             const float* __restrict__ cv,
                             float* __restrict__ keyOut,     // [NKV][SK][HD]
                             float* __restrict__ valOut,
                             bf16_t* __restrict__ Kb,        // [NKV][SK][HD]
                             bf16_t* __restrict__ Vt) {      // [NKV][HD][SK]
  size_t idx = (size_t)blockIdx.x * blockDim.x + threadIdx.x;
  if (idx >= (size_t)NKV * PAST * HD) return;
  int d = idx & (HD - 1);
  size_t rest = idx >> 7;
  int l = (int)(rest % PAST);
  int h = (int)(rest / PAST);
  float kv = ck[idx], vv = cv[idx];
  size_t o = ((size_t)h * SK + l) * HD + d;
  keyOut[o] = kv;
  valOut[o] = vv;
  Kb[o] = (bf16_t)kv;
  Vt[((size_t)h * HD + d) * SK + l] = (bf16_t)vv;
}

// New V (no RoPE): tuple output + transposed bf16.
__global__ void ingest_v_new(const float* __restrict__ Vf,   // [SQ][KVDIM]
                             float* __restrict__ valOut,
                             bf16_t* __restrict__ Vt) {
  size_t idx = (size_t)blockIdx.x * blockDim.x + threadIdx.x;
  if (idx >= (size_t)SQ * KVDIM) return;
  int d = idx & (HD - 1);
  int h = (int)((idx >> 7) & (NKV - 1));
  int s = (int)(idx >> 10);
  float vv = Vf[idx];
  size_t l = (size_t)PAST + s;
  valOut[((size_t)h * SK + l) * HD + d] = vv;
  Vt[((size_t)h * HD + d) * SK + l] = (bf16_t)vv;
}

// ---------------------------------------------------------------------------
// Flash attention.  Grid (SQ/128, NH), block 256 = 8 waves; each wave owns a
// 16-row Q tile and streams 32 keys/iter: 8 WMMAs scores + 8 WMMAs P*V.
// Score D-layout (lane = key column, VGPR r = q row) matches 16-lane
// shfl_xor row reductions under wave32.  P is transposed through a per-wave
// LDS tile back into A-fragment layout.
// ---------------------------------------------------------------------------
__global__ __launch_bounds__(256) void attn_flash(
    const bf16_t* __restrict__ Qb,   // [NH][SQ][HD]
    const bf16_t* __restrict__ Kb,   // [NKV][SK][HD]
    const bf16_t* __restrict__ Vt,   // [NKV][HD][SK]
    bf16_t* __restrict__ Ctxb) {     // [SQ][QDIM]
  __shared__ __align__(16) bf16_t Ps[8][16][40];

  int tid  = threadIdx.x;
  int lane = tid & 31;
  int w    = tid >> 5;
  int h    = blockIdx.y;
  int hk   = h >> 2;                       // G = NH/NKV = 4
  int q0   = blockIdx.x * 128 + w * 16;

  const bf16_t* Qh = Qb + (size_t)h  * SQ * HD;
  const bf16_t* Kh = Kb + (size_t)hk * SK * HD;
  const bf16_t* Vh = Vt + (size_t)hk * HD * SK;

  v16bf aq[4];
#pragma unroll
  for (int ds = 0; ds < 4; ++ds) aq[ds] = frag16(Qh, HD, q0, ds * 32, lane);

  v8f acc[8];
#pragma unroll
  for (int ni = 0; ni < 8; ++ni)
    acc[ni] = (v8f){0.f,0.f,0.f,0.f,0.f,0.f,0.f,0.f};
  float mrow[8], lrow[8];
#pragma unroll
  for (int r = 0; r < 8; ++r) { mrow[r] = -1e30f; lrow[r] = 0.f; }

  const float scale = 0.08838834764831845f;   // 1/sqrt(128)
  int ncol   = lane & 15;
  int ra     = (lane & 16) ? 8 : 0;
  int limit0 = PAST + q0;                      // causal limit for row 0
  int kend   = limit0 + 16;                    // <= SK by construction
  if (kend > SK) kend = SK;

  for (int kb = 0; kb < kend; kb += 32) {
    v8f s0 = (v8f){0.f,0.f,0.f,0.f,0.f,0.f,0.f,0.f};
    v8f s1 = (v8f){0.f,0.f,0.f,0.f,0.f,0.f,0.f,0.f};
#pragma unroll
    for (int ds = 0; ds < 4; ++ds) {
      v16bf k0f = frag16(Kh, HD, kb,      ds * 32, lane);
      v16bf k1f = frag16(Kh, HD, kb + 16, ds * 32, lane);
      s0 = wmma_bf16(aq[ds], k0f, s0);
      s1 = wmma_bf16(aq[ds], k1f, s1);
    }

    bool full = (kb + 31 <= limit0);
    float p0[8], p1[8];
#pragma unroll
    for (int r = 0; r < 8; ++r) {
      float a = s0[r] * scale, b = s1[r] * scale;
      if (!full) {
        int lim = limit0 + r + ra;
        if (kb + ncol      > lim) a = -1e30f;
        if (kb + 16 + ncol > lim) b = -1e30f;
      }
      p0[r] = a; p1[r] = b;
    }

#pragma unroll
    for (int r = 0; r < 8; ++r) {
      float v = fmaxf(p0[r], p1[r]);
      v = fmaxf(v, __shfl_xor(v, 1, 32));
      v = fmaxf(v, __shfl_xor(v, 2, 32));
      v = fmaxf(v, __shfl_xor(v, 4, 32));
      v = fmaxf(v, __shfl_xor(v, 8, 32));
      float nm    = fmaxf(mrow[r], v);
      float alpha = __expf(mrow[r] - nm);
      mrow[r] = nm;
      float e0 = __expf(p0[r] - nm);
      float e1 = __expf(p1[r] - nm);
      p0[r] = e0; p1[r] = e1;
      float rs = e0 + e1;
      rs += __shfl_xor(rs, 1, 32);
      rs += __shfl_xor(rs, 2, 32);
      rs += __shfl_xor(rs, 4, 32);
      rs += __shfl_xor(rs, 8, 32);
      lrow[r] = lrow[r] * alpha + rs;
#pragma unroll
      for (int ni = 0; ni < 8; ++ni) acc[ni][r] *= alpha;
    }

    // Transpose P (16q x 32k) into A-fragment layout via wave-private LDS.
#pragma unroll
    for (int r = 0; r < 8; ++r) {
      Ps[w][r + ra][ncol]      = (bf16_t)p0[r];
      Ps[w][r + ra][16 + ncol] = (bf16_t)p1[r];
    }
    v16bf pf = frag16(&Ps[w][0][0], 40, 0, 0, lane);

#pragma unroll
    for (int ni = 0; ni < 8; ++ni) {
      v16bf vf = frag16(Vh, SK, ni * 16, kb, lane);  // B col n = head-dim d
      acc[ni] = wmma_bf16(pf, vf, acc[ni]);
    }
  }

#pragma unroll
  for (int r = 0; r < 8; ++r) {
    float inv = 1.f / lrow[r];
    int gq = q0 + r + ra;
#pragma unroll
    for (int ni = 0; ni < 8; ++ni)
      Ctxb[(size_t)gq * QDIM + h * HD + ni * 16 + ncol] =
          (bf16_t)(acc[ni][r] * inv);
  }
}

// ---------------------------------------------------------------------------
extern "C" void kernel_launch(void* const* d_in, const int* in_sizes, int n_in,
                              void* d_out, int out_size, void* d_ws, size_t ws_size,
                              hipStream_t stream) {
  const float* X   = (const float*)d_in[0];
  const int*   pos = (const int*)  d_in[1];
  const float* ck  = (const float*)d_in[2];
  const float* cv  = (const float*)d_in[3];
  const float* Wq  = (const float*)d_in[4];
  const float* bq  = (const float*)d_in[5];
  const float* Wk  = (const float*)d_in[6];
  const float* bk  = (const float*)d_in[7];
  const float* Wv  = (const float*)d_in[8];
  const float* bv  = (const float*)d_in[9];
  const float* Wo  = (const float*)d_in[10];

  float* out    = (float*)d_out;                       // [SQ][QDIM]
  float* keyOut = out + (size_t)SQ * QDIM;             // [NKV][SK][HD]
  float* valOut = keyOut + (size_t)NKV * SK * HD;      // [NKV][SK][HD]

  // Workspace carve-up (~252 MB). Ctx bf16 aliases the dead X-bf16 region.
  char* ws = (char*)d_ws;
  size_t off = 0;
  auto carve = [&](size_t bytes) {
    char* p = ws + off;
    off += (bytes + 255) & ~(size_t)255;
    return p;
  };
  bf16_t* Xb  = (bf16_t*)carve((size_t)SQ * DIN * 2);      // 32 MB
  bf16_t* Wqb = (bf16_t*)carve((size_t)QDIM * DIN * 2);    // 64 MB
  bf16_t* Wkb = (bf16_t*)carve((size_t)KVDIM * DIN * 2);   // 16 MB
  bf16_t* Wvb = (bf16_t*)carve((size_t)KVDIM * DIN * 2);   // 16 MB
  bf16_t* Wob = (bf16_t*)carve((size_t)QDIM * QDIM * 2);   // 32 MB
  float*  Qf  = (float*) carve((size_t)SQ * QDIM * 4);     // 32 MB
  float*  Kf  = (float*) carve((size_t)SQ * KVDIM * 4);    //  8 MB
  float*  Vf  = (float*) carve((size_t)SQ * KVDIM * 4);    //  8 MB
  bf16_t* Qb  = (bf16_t*)carve((size_t)NH * SQ * HD * 2);  // 16 MB
  bf16_t* Kb  = (bf16_t*)carve((size_t)NKV * SK * HD * 2); //  8 MB
  bf16_t* Vt  = (bf16_t*)carve((size_t)NKV * HD * SK * 2); //  8 MB
  bf16_t* Ctxb = Xb;  // X is dead once the three projections finish

  auto cvt = [&](const float* in, bf16_t* o, size_t n) {
    f32_to_bf16<<<dim3((unsigned)((n / 8 + 255) / 256)), 256, 0, stream>>>(in, o, n);
  };
  cvt(X,  Xb,  (size_t)SQ * DIN);
  cvt(Wq, Wqb, (size_t)QDIM * DIN);
  cvt(Wk, Wkb, (size_t)KVDIM * DIN);
  cvt(Wv, Wvb, (size_t)KVDIM * DIN);
  cvt(Wo, Wob, (size_t)QDIM * QDIM);

  // Projections: C = X * W^T + b
  gemm_bf16<<<dim3(QDIM / 128, SQ / 128), 256, 0, stream>>>(
      Xb, Wqb, bq, Qf, SQ, QDIM, DIN);
  gemm_bf16<<<dim3(KVDIM / 128, SQ / 128), 256, 0, stream>>>(
      Xb, Wkb, bk, Kf, SQ, KVDIM, DIN);
  gemm_bf16<<<dim3(KVDIM / 128, SQ / 128), 256, 0, stream>>>(
      Xb, Wvb, bv, Vf, SQ, KVDIM, DIN);

  // KV tuple outputs + bf16 operand layouts
  {
    size_t n = (size_t)NKV * PAST * HD;
    ingest_cache<<<dim3((unsigned)((n + 255) / 256)), 256, 0, stream>>>(
        ck, cv, keyOut, valOut, Kb, Vt);
  }
  {
    size_t n = (size_t)SQ * (NH + NKV) * 64;
    rope_qk<<<dim3((unsigned)((n + 255) / 256)), 256, 0, stream>>>(
        Qf, Kf, pos, Qb, keyOut, Kb);
  }
  {
    size_t n = (size_t)SQ * KVDIM;
    ingest_v_new<<<dim3((unsigned)((n + 255) / 256)), 256, 0, stream>>>(
        Vf, valOut, Vt);
  }

  // Flash attention -> bf16 ctx [SQ][NH*HD]
  attn_flash<<<dim3(SQ / 128, NH), 256, 0, stream>>>(Qb, Kb, Vt, Ctxb);

  // Output projection: out = ctx * Wo^T
  gemm_bf16<<<dim3(QDIM / 128, SQ / 128), 256, 0, stream>>>(
      Ctxb, Wob, nullptr, out, SQ, QDIM, QDIM);
}
